// _MambaBlock_68942815035943
// MI455X (gfx1250) — matrix operand order repaired and checked by reading, not compiled, not run on hardware
//
#include <hip/hip_runtime.h>
#include <math.h>

// ---------------------------------------------------------------------------
// Mamba block for MI455X (gfx1250, wave32, WMMA bf16 16x16x32)
// B=2, L=1024, D_MODEL=1024, D_INNER=2048, D_STATE=16, DT_RANK=64, D_CONV=4
// ---------------------------------------------------------------------------

#define BB      2
#define LL      1024
#define DM      1024
#define DI      2048
#define DS      16
#define DTR     64
#define DCONV   4
#define RR      (BB * LL)        // 2048 rows (tokens)
#define NXD     (DTR + 2 * DS)   // 96

typedef __attribute__((ext_vector_type(16))) __bf16       bf16x16;
typedef __attribute__((ext_vector_type(8)))  float        f32x8;
typedef __attribute__((ext_vector_type(4)))  unsigned int u32x4;

union ABFrag {
    bf16x16      v;
    unsigned int u[8];
    u32x4        q[2];
};

// ---------------------------------------------------------------------------
// f32 -> bf16 repack (weights / activations)
// ---------------------------------------------------------------------------
__global__ __launch_bounds__(256) void cvt_bf16_kernel(const float* __restrict__ in,
                                                       __bf16* __restrict__ out, int n) {
    int i = blockIdx.x * 256 + threadIdx.x;
    if (i < n) out[i] = (__bf16)in[i];
}

// ---------------------------------------------------------------------------
// LayerNorm over D_MODEL, one block (8 waves) per token row, bf16 output
// ---------------------------------------------------------------------------
__global__ __launch_bounds__(256) void layernorm_kernel(const float* __restrict__ x,
                                                        const float* __restrict__ g,
                                                        const float* __restrict__ bta,
                                                        __bf16* __restrict__ out, int D) {
    const int row = blockIdx.x;
    const float* xr = x + (size_t)row * D;
    __shared__ float red[8];
    __shared__ float s_mu, s_rstd;

    float s = 0.f;
    for (int i = threadIdx.x; i < D; i += 256) s += xr[i];
#pragma unroll
    for (int o = 16; o > 0; o >>= 1) s += __shfl_down(s, o, 32);
    if ((threadIdx.x & 31) == 0) red[threadIdx.x >> 5] = s;
    __syncthreads();
    if (threadIdx.x == 0) {
        float t = 0.f;
#pragma unroll
        for (int w = 0; w < 8; ++w) t += red[w];
        s_mu = t / (float)D;
    }
    __syncthreads();
    const float mu = s_mu;

    float v = 0.f;
    for (int i = threadIdx.x; i < D; i += 256) {
        float dd = xr[i] - mu;
        v += dd * dd;
    }
#pragma unroll
    for (int o = 16; o > 0; o >>= 1) v += __shfl_down(v, o, 32);
    if ((threadIdx.x & 31) == 0) red[threadIdx.x >> 5] = v;
    __syncthreads();
    if (threadIdx.x == 0) {
        float t = 0.f;
#pragma unroll
        for (int w = 0; w < 8; ++w) t += red[w];
        s_rstd = rsqrtf(t / (float)D + 1e-5f);
    }
    __syncthreads();
    const float rs = s_rstd;
    for (int i = threadIdx.x; i < D; i += 256)
        out[(size_t)row * D + i] = (__bf16)((xr[i] - mu) * rs * g[i] + bta[i]);
}

// ---------------------------------------------------------------------------
// WMMA GEMM:  O[M,N] = A[M,K](bf16,row-major) * Bt[N,K](bf16,row-major)^T
//                      + bias[N] (opt) + resid[M,N] (opt)
//
// Block = 256 threads = 8 waves arranged 2(M) x 4(N); each wave register-
// blocks a 32x32 output tile (2x2 accumulators) -> 4 WMMAs per k-step on
// 4 fragments (16 FLOP/byte of operand traffic vs 8 for 1x1 blocking).
// Block tile = 64 x 128.  Requires M % 64 == 0, N % 32 == 0 (holds for all
// call sites: N in {4096, 96, 2048, 1024}).
//
// Fragment layouts per CDNA5 ISA 7.12.2 (wave32):
//   A 16x32 : lane m=l&15, half=l>>4; vgpr p holds K pair; pairs p0..3 and
//             p4..7 are contiguous 16B runs at (8h) and (8h+16) elements.
//   B 32x16 : lane n=l&15, half=l>>4; same K order as A (16B runs at 8h, 8h+16).
//   C/D     : vgpr v -> row v+8h, col l&15
// ---------------------------------------------------------------------------
__global__ __launch_bounds__(256) void wmma_gemm_bf16_kernel(
    const __bf16* __restrict__ A, const __bf16* __restrict__ Bt,
    const float* __restrict__ bias, const float* __restrict__ resid,
    float* __restrict__ O, int M, int N, int K) {
    const int lane = threadIdx.x & 31;
    const int wave = threadIdx.x >> 5;
    const int wm   = wave >> 2;               // 0..1
    const int wn   = wave & 3;                // 0..3
    const int tileM = blockIdx.y * 64 + wm * 32;
    const int tileN = blockIdx.x * 128 + wn * 32;
    if (tileN >= N) return;  // wave-uniform: EXEC stays all-ones for WMMA

    const int mr   = lane & 15;
    const int half = lane >> 4;

    const __bf16* aRow0 = A + (size_t)(tileM + mr) * K + (half << 3);
    const __bf16* aRow1 = aRow0 + (size_t)16 * K;
    const __bf16* bRow0 = Bt + (size_t)(tileN + mr) * K + (half << 3);
    const __bf16* bRow1 = bRow0 + (size_t)16 * K;

    f32x8 acc00 = {}, acc01 = {}, acc10 = {}, acc11 = {};
#pragma unroll 2
    for (int k0 = 0; k0 < K; k0 += 32) {
        ABFrag a0, a1, b0, b1;
        // A fragments: pairs p0..3 at +0 (16B), pairs p4..7 at +16 elems (16B)
        a0.q[0] = *(const u32x4*)(aRow0 + k0);
        a0.q[1] = *(const u32x4*)(aRow0 + k0 + 16);
        a1.q[0] = *(const u32x4*)(aRow1 + k0);
        a1.q[1] = *(const u32x4*)(aRow1 + k0 + 16);
        // B fragments: same layout as A: 16B at +0, 16B at +16 elems
        b0.q[0] = *(const u32x4*)(bRow0 + k0);
        b0.q[1] = *(const u32x4*)(bRow0 + k0 + 16);
        b1.q[0] = *(const u32x4*)(bRow1 + k0);
        b1.q[1] = *(const u32x4*)(bRow1 + k0 + 16);
        if (k0 + 32 < K) {
            __builtin_prefetch(aRow0 + k0 + 32, 0, 1);
            __builtin_prefetch(aRow1 + k0 + 32, 0, 1);
            __builtin_prefetch(bRow0 + k0 + 32, 0, 1);
            __builtin_prefetch(bRow1 + k0 + 32, 0, 1);
        }
        acc00 = __builtin_amdgcn_wmma_f32_16x16x32_bf16(false, a0.v, false, b0.v,
                                                        (short)0, acc00, false, false);
        acc01 = __builtin_amdgcn_wmma_f32_16x16x32_bf16(false, a0.v, false, b1.v,
                                                        (short)0, acc01, false, false);
        acc10 = __builtin_amdgcn_wmma_f32_16x16x32_bf16(false, a1.v, false, b0.v,
                                                        (short)0, acc10, false, false);
        acc11 = __builtin_amdgcn_wmma_f32_16x16x32_bf16(false, a1.v, false, b1.v,
                                                        (short)0, acc11, false, false);
    }

    const float bval0 = bias ? bias[tileN + mr]      : 0.0f;
    const float bval1 = bias ? bias[tileN + 16 + mr] : 0.0f;
#pragma unroll
    for (int v = 0; v < 8; ++v) {
        int row0 = tileM + v + (half << 3);
        int row1 = row0 + 16;
        size_t i00 = (size_t)row0 * N + (size_t)(tileN + mr);
        size_t i01 = i00 + 16;
        size_t i10 = (size_t)row1 * N + (size_t)(tileN + mr);
        size_t i11 = i10 + 16;
        float r00 = resid ? resid[i00] : 0.0f;
        float r01 = resid ? resid[i01] : 0.0f;
        float r10 = resid ? resid[i10] : 0.0f;
        float r11 = resid ? resid[i11] : 0.0f;
        O[i00] = acc00[v] + bval0 + r00;
        O[i01] = acc01[v] + bval1 + r01;
        O[i10] = acc10[v] + bval0 + r10;
        O[i11] = acc11[v] + bval1 + r11;
    }
}

// ---------------------------------------------------------------------------
// Depthwise causal conv (k=4) + SiLU over x-branch of xz; dual f32/bf16 output
// ---------------------------------------------------------------------------
__global__ __launch_bounds__(256) void conv_silu_kernel(const float* __restrict__ xz,
                                                        const float* __restrict__ cw,
                                                        const float* __restrict__ cb,
                                                        float* __restrict__ xc,
                                                        __bf16* __restrict__ xc_bf) {
    int idx = blockIdx.x * 256 + threadIdx.x;  // over RR*DI
    if (idx >= RR * DI) return;
    int c = idx & (DI - 1);
    int r = idx >> 11;       // / DI
    int l = r & (LL - 1);
    float acc = cb[c];
#pragma unroll
    for (int k = 0; k < DCONV; ++k) {
        int lk = l - (DCONV - 1) + k;
        if (lk >= 0)
            acc += xz[(size_t)(r - (DCONV - 1) + k) * (2 * DI) + c] * cw[c * DCONV + k];
    }
    float s = acc / (1.0f + __expf(-acc));  // SiLU
    xc[idx]    = s;
    xc_bf[idx] = (__bf16)s;
}

// ---------------------------------------------------------------------------
// Extract dt_lo columns (0..63) of x_dbl as bf16 for the dt GEMM
// ---------------------------------------------------------------------------
__global__ __launch_bounds__(256) void extract_dtlo_kernel(const float* __restrict__ xdbl,
                                                           __bf16* __restrict__ dtlo) {
    int i = blockIdx.x * 256 + threadIdx.x;  // over RR*DTR
    if (i >= RR * DTR) return;
    int r = i >> 6, j = i & 63;
    dtlo[i] = (__bf16)xdbl[r * NXD + j];
}

// ---------------------------------------------------------------------------
// Selective scan: one thread per (b, d); h[16] in registers; B_s/C_s rows
// staged through LDS in chunks of 32 timesteps. Fuses softplus(dt), D*x_c,
// SiLU(z) gate; emits bf16 y for the output GEMM.
// ---------------------------------------------------------------------------
#define SCAN_CHUNK 32
__global__ __launch_bounds__(256) void scan_kernel(const float* __restrict__ dt_raw,
                                                   const float* __restrict__ xc,
                                                   const float* __restrict__ xdbl,
                                                   const float* __restrict__ xz,
                                                   const float* __restrict__ A_log,
                                                   const float* __restrict__ Dp,
                                                   __bf16* __restrict__ y_bf) {
    const int b = blockIdx.y;
    const int d = blockIdx.x * 256 + threadIdx.x;

    float Aa[DS];
#pragma unroll
    for (int s = 0; s < DS; ++s) Aa[s] = -__expf(A_log[d * DS + s]);
    float h[DS];
#pragma unroll
    for (int s = 0; s < DS; ++s) h[s] = 0.f;
    const float Dd = Dp[d];

    __shared__ float Bsh[SCAN_CHUNK][DS];
    __shared__ float Csh[SCAN_CHUNK][DS];

    for (int l0 = 0; l0 < LL; l0 += SCAN_CHUNK) {
        __syncthreads();
        for (int t = threadIdx.x; t < SCAN_CHUNK * DS; t += 256) {
            int ll = t >> 4, ss = t & 15;
            int row = b * LL + l0 + ll;
            Bsh[ll][ss] = xdbl[(size_t)row * NXD + DTR + ss];
            Csh[ll][ss] = xdbl[(size_t)row * NXD + DTR + DS + ss];
        }
        __syncthreads();
        for (int li = 0; li < SCAN_CHUNK; ++li) {
            int row = b * LL + l0 + li;
            float dtv = dt_raw[(size_t)row * DI + d];
            dtv = (dtv > 20.f) ? dtv : log1pf(__expf(dtv));  // softplus
            float xcv = xc[(size_t)row * DI + d];
            float du  = dtv * xcv;
            float y = 0.f;
#pragma unroll
            for (int s = 0; s < DS; ++s) {
                h[s] = __expf(dtv * Aa[s]) * h[s] + du * Bsh[li][s];
                y += h[s] * Csh[li][s];
            }
            y += Dd * xcv;
            float zv = xz[(size_t)row * (2 * DI) + DI + d];
            float yg = y * (zv / (1.0f + __expf(-zv)));  // y * silu(z)
            y_bf[(size_t)row * DI + d] = (__bf16)yg;
        }
    }
}

// ---------------------------------------------------------------------------
// Host-side launch sequencing
// ---------------------------------------------------------------------------
extern "C" void kernel_launch(void* const* d_in, const int* in_sizes, int n_in,
                              void* d_out, int out_size, void* d_ws, size_t ws_size,
                              hipStream_t stream) {
    (void)in_sizes; (void)n_in; (void)out_size; (void)ws_size;

    const float* x      = (const float*)d_in[0];
    const float* ln_g   = (const float*)d_in[1];
    const float* ln_b   = (const float*)d_in[2];
    const float* W_in   = (const float*)d_in[3];
    const float* b_in   = (const float*)d_in[4];
    const float* conv_w = (const float*)d_in[5];
    const float* conv_b = (const float*)d_in[6];
    const float* W_x    = (const float*)d_in[7];
    const float* W_dt   = (const float*)d_in[8];
    const float* b_dt   = (const float*)d_in[9];
    const float* A_log  = (const float*)d_in[10];
    const float* Dp     = (const float*)d_in[11];
    const float* W_out  = (const float*)d_in[12];
    const float* b_out  = (const float*)d_in[13];
    float* out = (float*)d_out;

    // workspace carve-up (256B aligned)
    char* ws = (char*)d_ws;
    size_t off = 0;
    auto carve = [&](size_t bytes) -> char* {
        char* p = ws + off;
        off += (bytes + 255) & ~(size_t)255;
        return p;
    };
    __bf16* xn_bf   = (__bf16*)carve((size_t)RR * DM * 2);
    __bf16* Win_bf  = (__bf16*)carve((size_t)(2 * DI) * DM * 2);
    __bf16* Wx_bf   = (__bf16*)carve((size_t)NXD * DI * 2);
    __bf16* Wdt_bf  = (__bf16*)carve((size_t)DI * DTR * 2);
    __bf16* Wout_bf = (__bf16*)carve((size_t)DM * DI * 2);
    float*  xz      = (float*)carve((size_t)RR * (2 * DI) * 4);
    float*  xc      = (float*)carve((size_t)RR * DI * 4);
    __bf16* xc_bf   = (__bf16*)carve((size_t)RR * DI * 2);
    float*  xdbl    = (float*)carve((size_t)RR * NXD * 4);
    __bf16* dtlo_bf = (__bf16*)carve((size_t)RR * DTR * 2);
    float*  dt_raw  = (float*)carve((size_t)RR * DI * 4);
    __bf16* y_bf    = (__bf16*)carve((size_t)RR * DI * 2);

    auto cvt = [&](const float* src, __bf16* dst, int n) {
        cvt_bf16_kernel<<<(n + 255) / 256, 256, 0, stream>>>(src, dst, n);
    };

    // 1) weight repacks to bf16 (cheap; rerun each call for determinism)
    cvt(W_in,  Win_bf,  2 * DI * DM);
    cvt(W_x,   Wx_bf,   NXD * DI);
    cvt(W_dt,  Wdt_bf,  DI * DTR);
    cvt(W_out, Wout_bf, DM * DI);

    // 2) LayerNorm -> bf16 xn
    layernorm_kernel<<<RR, 256, 0, stream>>>(x, ln_g, ln_b, xn_bf, DM);

    // 3) xz = xn @ W_in^T + b_in   [2048 x 4096], K=1024
    {
        dim3 grid((2 * DI + 127) / 128, RR / 64);
        wmma_gemm_bf16_kernel<<<grid, 256, 0, stream>>>(xn_bf, Win_bf, b_in, nullptr,
                                                        xz, RR, 2 * DI, DM);
    }

    // 4) depthwise causal conv + SiLU on x-branch
    conv_silu_kernel<<<(RR * DI + 255) / 256, 256, 0, stream>>>(xz, conv_w, conv_b, xc, xc_bf);

    // 5) x_dbl = x_c @ W_x^T       [2048 x 96], K=2048
    {
        dim3 grid((NXD + 127) / 128, RR / 64);
        wmma_gemm_bf16_kernel<<<grid, 256, 0, stream>>>(xc_bf, Wx_bf, nullptr, nullptr,
                                                        xdbl, RR, NXD, DI);
    }

    // 6) dt_lo slice -> bf16
    extract_dtlo_kernel<<<(RR * DTR + 255) / 256, 256, 0, stream>>>(xdbl, dtlo_bf);

    // 7) dt_raw = dt_lo @ W_dt^T + b_dt   [2048 x 2048], K=64 (softplus in scan)
    {
        dim3 grid((DI + 127) / 128, RR / 64);
        wmma_gemm_bf16_kernel<<<grid, 256, 0, stream>>>(dtlo_bf, Wdt_bf, b_dt, nullptr,
                                                        dt_raw, RR, DI, DTR);
    }

    // 8) selective scan + D*x_c + SiLU(z) gate -> bf16 y
    {
        dim3 grid(DI / 256, BB);
        scan_kernel<<<grid, 256, 0, stream>>>(dt_raw, xc, xdbl, xz, A_log, Dp, y_bf);
    }

    // 9) out = x + y @ W_out^T + b_out    [2048 x 1024], K=2048 (residual fused)
    {
        dim3 grid((DM + 127) / 128, RR / 64);
        wmma_gemm_bf16_kernel<<<grid, 256, 0, stream>>>(y_bf, Wout_bf, b_out, x,
                                                        out, RR, DM, DI);
    }
}
